// VectorQuantizer_62294205662007
// MI455X (gfx1250) — matrix operand order repaired and checked by reading, not compiled, not run on hardware
//
#include <hip/hip_runtime.h>
#include <hip/hip_bf16.h>

typedef __attribute__((ext_vector_type(16))) _Float16 v16h;
typedef __attribute__((ext_vector_type(8)))  float    v8f;

#define EMBED_NUM 8192
#define EMBED_DIM 64
#define N_ROWS    32768          // 8*64*64
#define Z_ELEMS   2097152        // 8*64*64*64
#define CHUNK     128            // codebook entries staged in LDS per iteration

// d_out layout (floats):
//   [0, 2097152)                : z_q in [b,c,h,w]
//   [2097152]                   : loss scalar
//   [2097153, 2097153+32768)    : encoding indices (as float)
//   [2129921, 2129921+8192)     : bin_count (as float)
#define OUT_LOSS  2097152
#define OUT_IDX   2097153
#define OUT_BIN   2129921

union H2 { _Float16 h[2]; unsigned int u; };

// ---------------------------------------------------------------------------
// Prep: normalize codebook -> f16 in workspace; zero bincount + loss accum.
// One wave (32 lanes) per codebook row; lane handles dims {2l, 2l+1}.
// ---------------------------------------------------------------------------
__global__ void vq_prep_kernel(const float* __restrict__ embed,
                               _Float16* __restrict__ cb16,
                               float* __restrict__ out_bin,
                               float* __restrict__ loss_acc) {
    const int lane = threadIdx.x & 31;
    const int row  = blockIdx.x * (blockDim.x >> 5) + (threadIdx.x >> 5);
    if (row < EMBED_NUM) {
        const float* r = embed + (size_t)row * EMBED_DIM;
        float x0 = r[2 * lane], x1 = r[2 * lane + 1];
        float ss = x0 * x0 + x1 * x1;
        #pragma unroll
        for (int s = 1; s < 32; s <<= 1) ss += __shfl_xor(ss, s, 32);
        float inv = 1.0f / fmaxf(sqrtf(ss), 1e-12f);
        H2 p;
        p.h[0] = (_Float16)(x0 * inv);
        p.h[1] = (_Float16)(x1 * inv);
        ((unsigned int*)cb16)[row * 32 + lane] = p.u;
    }
    const int gid = blockIdx.x * blockDim.x + threadIdx.x;
    if (gid < EMBED_NUM) out_bin[gid] = 0.0f;
    if (gid == 0) loss_acc[0] = 0.0f;
}

// ---------------------------------------------------------------------------
// Main fused kernel: normalize z rows into WMMA A-fragments (registers),
// sweep codebook in LDS chunks, v_wmma_f32_16x16x32_f16 + running argmax,
// then gather/scatter z_q, indices, bincount, loss partial sums.
// Block = 256 threads (8 wave32), each wave owns 16 z rows -> 128 rows/block.
// ---------------------------------------------------------------------------
__global__ void __launch_bounds__(256)
vq_main_kernel(const float* __restrict__ z,
               const float* __restrict__ embed,
               const _Float16* __restrict__ cb16,
               float* __restrict__ out_zq,
               float* __restrict__ out_idx,
               float* __restrict__ out_bin,
               float* __restrict__ loss_acc) {
    __shared__ __align__(32) _Float16 lds_cb[CHUNK * EMBED_DIM];  // 16 KB

    const int lane = threadIdx.x & 31;
    const int wave = threadIdx.x >> 5;
    const int m    = lane & 15;       // A-matrix row within wave tile
    const int half = lane >> 4;       // lane group (K-split per ISA layout)

    // ---- Load + normalize this lane's share of its z row into A fragments.
    // Lane (half=0): K in {0..15, 32..47}
    // Lane (half=1): K in {16..31, 48..63}
    const int n0  = blockIdx.x * 128 + wave * 16 + m;
    const int b0  = n0 >> 12;
    const int hw0 = n0 & 4095;
    const float* zrow = z + (size_t)b0 * 262144 + hw0;  // stride 4096 per c

    float zv[32];
    float ss = 0.0f;
    #pragma unroll
    for (int g = 0; g < 4; ++g) {
        const int cb0 = (g >> 1) * 32 + half * 16 + (g & 1) * 8;
        #pragma unroll
        for (int j = 0; j < 8; ++j) {
            float v = zrow[(size_t)(cb0 + j) * 4096];
            zv[g * 8 + j] = v;
            ss += v * v;
        }
    }
    ss += __shfl_xor(ss, 16, 32);                 // pair lanes L and L+16
    const float inv = 1.0f / fmaxf(sqrtf(ss), 1e-12f);

    v16h a0, a1;
    #pragma unroll
    for (int j = 0; j < 16; ++j) {
        a0[j] = (_Float16)(zv[j]      * inv);     // K = 0..31 portion
        a1[j] = (_Float16)(zv[16 + j] * inv);     // K = 32..63 portion
    }

    // ---- Running argmax state: lane holds C-layout column N=lane&15,
    //      rows r + 8*half (r = 0..7).
    float best[8];
    int   bidx[8];
    #pragma unroll
    for (int r = 0; r < 8; ++r) { best[r] = -3.402823466e38f; bidx[r] = 0; }

    // ---- Codebook sweep.
    for (int cb = 0; cb < EMBED_NUM; cb += CHUNK) {
        // stage chunk into LDS (128x64 f16 = 16KB; 1024 x 16B copies)
        const uint4* s4 = (const uint4*)(cb16 + (size_t)cb * EMBED_DIM);
        uint4* d4 = (uint4*)lds_cb;
        for (int i = threadIdx.x; i < (CHUNK * EMBED_DIM * 2) / 16; i += 256)
            d4[i] = s4[i];
        // prefetch next chunk into cache while we compute
        if (cb + CHUNK < EMBED_NUM) {
            const char* nxt = (const char*)(cb16 + (size_t)(cb + CHUNK) * EMBED_DIM);
            __builtin_prefetch(nxt + (size_t)threadIdx.x * 64, 0, 0);
        }
        __syncthreads();

        #pragma unroll
        for (int t = 0; t < CHUNK / 16; ++t) {
            // B fragments: lane group half=0 -> K 0..15 (+32), half=1 -> K 16..31 (+48)
            const _Float16* bp = &lds_cb[(t * 16 + m) * EMBED_DIM + half * 16];
            v16h bf0 = *(const v16h*)bp;          // K = half*16 .. half*16+15
            v16h bf1 = *(const v16h*)(bp + 32);   // K = 32 + half*16 .. +15

            v8f acc = {};
            acc = __builtin_amdgcn_wmma_f32_16x16x32_f16(
                false, a0, false, bf0, (short)0, acc, false, false);
            acc = __builtin_amdgcn_wmma_f32_16x16x32_f16(
                false, a1, false, bf1, (short)0, acc, false, false);

            const int code = cb + t * 16 + m;     // column this lane owns
            #pragma unroll
            for (int r = 0; r < 8; ++r) {
                float v = acc[r];
                if (v > best[r]) { best[r] = v; bidx[r] = code; }
            }
        }
        __syncthreads();
    }

    // ---- Per-row argmax across the 16 columns of each half-group,
    //      then gather embed row, scatter z_q, indices, bincount, loss.
    float lsum = 0.0f;
    const int lg = lane & 15;
    #pragma unroll
    for (int r = 0; r < 8; ++r) {
        float v = best[r];
        int   bi = bidx[r];
        #pragma unroll
        for (int s = 1; s < 16; s <<= 1) {
            float ov = __shfl_xor(v, s, 16);
            int   oi = __shfl_xor(bi, s, 16);
            if (ov > v || (ov == v && oi < bi)) { v = ov; bi = oi; }
        }
        const int rowm = r + 8 * half;                       // C-layout row
        const int n2   = blockIdx.x * 128 + wave * 16 + rowm;
        const int b2   = n2 >> 12;
        const int hw2  = n2 & 4095;

        // cooperative: 16 lanes of the group each handle 4 consecutive c
        const float4 e = *(const float4*)(embed + (size_t)bi * EMBED_DIM + lg * 4);
        float*       oz = out_zq + (size_t)b2 * 262144 + hw2;
        const float* zr = z      + (size_t)b2 * 262144 + hw2;
        const int c0 = lg * 4;
        float d0 = e.x - zr[(size_t)(c0 + 0) * 4096];
        float d1 = e.y - zr[(size_t)(c0 + 1) * 4096];
        float d2 = e.z - zr[(size_t)(c0 + 2) * 4096];
        float d3 = e.w - zr[(size_t)(c0 + 3) * 4096];
        oz[(size_t)(c0 + 0) * 4096] = e.x;
        oz[(size_t)(c0 + 1) * 4096] = e.y;
        oz[(size_t)(c0 + 2) * 4096] = e.z;
        oz[(size_t)(c0 + 3) * 4096] = e.w;
        lsum += d0 * d0 + d1 * d1 + d2 * d2 + d3 * d3;

        if (lg == 0) {
            out_idx[n2] = (float)bi;
            atomicAdd(&out_bin[bi], 1.0f);
        }
    }
    #pragma unroll
    for (int s = 1; s < 32; s <<= 1) lsum += __shfl_xor(lsum, s, 32);
    if (lane == 0) atomicAdd(loss_acc, lsum);
}

// ---------------------------------------------------------------------------
// Finalize: loss = (BETA + 1) * mean = 1.25 * sum / 2097152
// ---------------------------------------------------------------------------
__global__ void vq_finalize_kernel(const float* __restrict__ loss_acc,
                                   float* __restrict__ out_loss) {
    if (threadIdx.x == 0 && blockIdx.x == 0)
        out_loss[0] = 1.25f * loss_acc[0] * (1.0f / 2097152.0f);
}

extern "C" void kernel_launch(void* const* d_in, const int* in_sizes, int n_in,
                              void* d_out, int out_size, void* d_ws, size_t ws_size,
                              hipStream_t stream) {
    const float* z     = (const float*)d_in[0];   // [8,64,64,64]
    const float* embed = (const float*)d_in[1];   // [8192,64]

    float* out = (float*)d_out;

    // workspace: [0, 1MB) f16 normalized codebook; [1MB] loss accumulator
    _Float16* cb16     = (_Float16*)d_ws;
    float*    loss_acc = (float*)((char*)d_ws + (size_t)EMBED_NUM * EMBED_DIM * 2);

    // prep: 8192 rows, 1 wave each -> 8 rows/block, 1024 blocks
    vq_prep_kernel<<<1024, 256, 0, stream>>>(embed, cb16, out + OUT_BIN, loss_acc);

    // main: 32768 rows / 128 per block = 256 blocks
    vq_main_kernel<<<256, 256, 0, stream>>>(z, embed, cb16,
                                            out, out + OUT_IDX, out + OUT_BIN,
                                            loss_acc);

    vq_finalize_kernel<<<1, 32, 0, stream>>>(loss_acc, out + OUT_LOSS);
}
